// CILRS_266287972950
// MI455X (gfx1250) — matrix operand order repaired and checked by reading, not compiled, not run on hardware
//
#include <hip/hip_runtime.h>

#define NB 32768  // batch

typedef __attribute__((ext_vector_type(16))) _Float16 v16h;
typedef __attribute__((ext_vector_type(8)))  _Float16 v8h;
typedef __attribute__((ext_vector_type(8)))  float    v8f;

// ---------------- packing / conversion kernels ----------------

__global__ __launch_bounds__(256) void k_cvt_f16(const float* __restrict__ s,
                                                 _Float16* __restrict__ d, int n) {
  int i = blockIdx.x * 256 + threadIdx.x;
  if (i < n) d[i] = (_Float16)s[i];
}

// s: [E][K,N] f32 row-major  ->  d: [E][N,K] f16 row-major (B^T so WMMA B-frag is contiguous)
__global__ __launch_bounds__(256) void k_transpose_f16(const float* __restrict__ s,
                                                       _Float16* __restrict__ d,
                                                       int K, int N) {
  size_t eoff = (size_t)blockIdx.y * K * N;
  int i = blockIdx.x * 256 + threadIdx.x;
  if (i < K * N) {
    int k = i / N, n = i - k * N;
    d[eoff + (size_t)n * K + k] = (_Float16)s[eoff + i];
  }
}

// sp layer1: h1[row,j] = relu(speed[row]*W1[j] + b1[j])   (K=1 outer product)
__global__ __launch_bounds__(256) void k_sp_layer1(const float* __restrict__ speed,
                                                   const float* __restrict__ W1,
                                                   const float* __restrict__ b1,
                                                   _Float16* __restrict__ out) {
  int i = blockIdx.x * 256 + threadIdx.x;   // NB*128 threads
  int row = i >> 7, j = i & 127;
  float v = fmaxf(fmaf(speed[row], W1[j], b1[j]), 0.f);
  out[i] = (_Float16)v;
}

// ---------------- generic WMMA f16 GEMM ----------------
// Out[e][M,N] = act( (A[e][M,Ka] ++ A2[M,Kb]) @ B[e][K,N] + bias[e][N] )
// Bt is B transposed: [E][N,K] f16.
// Block = 8 waves; wave tile = 32 rows x 64 cols (2 row-tiles x 4 col-tiles,
// 8 WMMAs per 32-deep k-step). K-loop is unrolled x2 with ping-pong fragment
// buffers: next step's fragments prefetch while the current step's WMMAs
// issue, and no register rotation is needed between steps.
// All K used here are multiples of 64.
__global__ __launch_bounds__(256) void k_gemm_wmma(
    const _Float16* __restrict__ A, size_t a_estride, int Ka,
    const _Float16* __restrict__ A2, int Kb,
    const _Float16* __restrict__ Bt,
    const float* __restrict__ bias,
    _Float16* __restrict__ Out,
    int M, int N, int relu)
{
  const int K = Ka + Kb;
  const int e = blockIdx.z;
  A    += (size_t)e * a_estride;
  Bt   += (size_t)e * N * K;
  bias += (size_t)e * N;
  Out  += (size_t)e * M * N;

  const int wave = threadIdx.x >> 5;
  const int lane = threadIdx.x & 31;
  const int lm = lane & 15;   // A-row / B-col / C-col within 16x16 tile
  const int lh = lane >> 4;   // half selector
  const int m0 = blockIdx.x * 256 + wave * 32;   // two 16-row tiles
  const int n0 = blockIdx.y * 64;

  // Hoisted per-lane base pointers (k advances by constant element offsets).
  const _Float16* bp[4];
#pragma unroll
  for (int t = 0; t < 4; ++t)
    bp[t] = Bt + (size_t)(n0 + 16 * t + lm) * K + lh * 8;

  const _Float16* apA[2];
  const _Float16* apA2[2];
#pragma unroll
  for (int r = 0; r < 2; ++r) {
    apA[r]  = A + (size_t)(m0 + 16 * r + lm) * Ka + lh * 8;
    apA2[r] = A2 ? A2 + (size_t)(m0 + 16 * r + lm) * Kb + lh * 8 : (const _Float16*)0;
  }

  // A fragment (16-bit 16x32 layout): lanes 0-15 hold K {0..7,16..23},
  // lanes 16-31 hold K {8..15,24..31} -> two b128 loads, second at +16 elems.
  auto ldA = [&](const _Float16* p) {
    v8h lo = *(const v8h*)p;
    v8h hi = *(const v8h*)(p + 16);
    v16h f;
#pragma unroll
    for (int i = 0; i < 8; ++i) { f[i] = lo[i]; f[i + 8] = hi[i]; }
    return f;
  };
  // B fragment from Bt[N,K]: same K layout as A -> two b128 loads, second at +16 elems.
  auto ldB = [&](const _Float16* p) {
    v8h lo = *(const v8h*)p;
    v8h hi = *(const v8h*)(p + 16);
    v16h f;
#pragma unroll
    for (int i = 0; i < 8; ++i) { f[i] = lo[i]; f[i + 8] = hi[i]; }
    return f;
  };
  // split-A: k < Ka -> A, else A2 (join layer: p_i ++ v_emb)
  auto aPtr = [&](int r, int k0) {
    return (k0 < Ka) ? apA[r] + k0 : apA2[r] + (k0 - Ka);
  };

  v8f acc[2][4] = {};

  v16h ca[2], cb[4];   // ping buffer
  v16h da[2], db[4];   // pong buffer
#pragma unroll
  for (int r = 0; r < 2; ++r) ca[r] = ldA(aPtr(r, 0));
#pragma unroll
  for (int t = 0; t < 4; ++t) cb[t] = ldB(bp[t]);

  auto mma8 = [&](v16h (&fa)[2], v16h (&fb)[4]) {
#pragma unroll
    for (int r = 0; r < 2; ++r)
#pragma unroll
      for (int t = 0; t < 4; ++t)
        acc[r][t] = __builtin_amdgcn_wmma_f32_16x16x32_f16(
            false, fa[r], false, fb[t], (short)0, acc[r][t], false, false);
  };

  for (int k0 = 0; k0 < K; k0 += 64) {
    if (k0 + 32 < K) {   // prefetch pong while ping computes
#pragma unroll
      for (int r = 0; r < 2; ++r) da[r] = ldA(aPtr(r, k0 + 32));
#pragma unroll
      for (int t = 0; t < 4; ++t) db[t] = ldB(bp[t] + (k0 + 32));
    }
    mma8(ca, cb);
    if (k0 + 64 < K) {   // prefetch ping while pong computes
#pragma unroll
      for (int r = 0; r < 2; ++r) ca[r] = ldA(aPtr(r, k0 + 64));
#pragma unroll
      for (int t = 0; t < 4; ++t) cb[t] = ldB(bp[t] + (k0 + 64));
    }
    if (k0 + 32 < K) mma8(da, db);
  }

  // epilogue: C/D layout -> VGPR q holds row (lh*8 + q), col = lm
#pragma unroll
  for (int r = 0; r < 2; ++r)
#pragma unroll
    for (int t = 0; t < 4; ++t) {
      int n = n0 + 16 * t + lm;
      float bn = bias[n];
#pragma unroll
      for (int q = 0; q < 8; ++q) {
        int m = m0 + 16 * r + lh * 8 + q;
        float v = acc[r][t][q] + bn;
        if (relu) v = fmaxf(v, 0.f);
        Out[(size_t)m * N + n] = (_Float16)v;
      }
    }
}

// ---------------- heads ----------------

// v_p[row] = dot(h[row,0:128], W3[0:128]) + b3   (wave per row)
__global__ __launch_bounds__(256) void k_vp_head(const _Float16* __restrict__ h,
                                                 const float* __restrict__ W3,
                                                 const float* __restrict__ b3,
                                                 float* __restrict__ out) {
  int wave = threadIdx.x >> 5, lane = threadIdx.x & 31;
  int row = blockIdx.x * 8 + wave;
  const _Float16* hp = h + (size_t)row * 128 + lane * 4;
  const float* wp = W3 + lane * 4;
  float acc = 0.f;
#pragma unroll
  for (int i = 0; i < 4; ++i) acc += (float)hp[i] * wp[i];
  for (int off = 16; off > 0; off >>= 1) acc += __shfl_down(acc, off, 32);
  if (lane == 0) out[row] = acc + b3[0];
}

// action[row,:] = h2[cmd][row,:] @ W3[cmd] + b3[cmd]  (wave per row; selected expert only)
__global__ __launch_bounds__(256) void k_action_head(const _Float16* __restrict__ ch2,
                                                     const int* __restrict__ command,
                                                     const float* __restrict__ W3,
                                                     const float* __restrict__ b3,
                                                     float* __restrict__ out) {
  int wave = threadIdx.x >> 5, lane = threadIdx.x & 31;
  int row = blockIdx.x * 8 + wave;
  int e = command[row] & 3;
  const _Float16* hp = ch2 + ((size_t)e * NB + row) * 256;
  const float* w = W3 + (size_t)e * 256 * 3;
  float a0 = 0.f, a1 = 0.f, a2 = 0.f;
  for (int k = lane; k < 256; k += 32) {
    float hv = (float)hp[k];
    a0 += hv * w[k * 3 + 0];
    a1 += hv * w[k * 3 + 1];
    a2 += hv * w[k * 3 + 2];
  }
  for (int off = 16; off > 0; off >>= 1) {
    a0 += __shfl_down(a0, off, 32);
    a1 += __shfl_down(a1, off, 32);
    a2 += __shfl_down(a2, off, 32);
  }
  if (lane == 0) {
    const float* bb = b3 + e * 3;
    float* o = out + (size_t)row * 3;
    o[0] = a0 + bb[0]; o[1] = a1 + bb[1]; o[2] = a2 + bb[2];
  }
}

// ---------------- host ----------------

extern "C" void kernel_launch(void* const* d_in, const int* in_sizes, int n_in,
                              void* d_out, int out_size, void* d_ws, size_t ws_size,
                              hipStream_t stream) {
  const int B = NB;
  const float* p_i    = (const float*)d_in[0];
  const float* speed  = (const float*)d_in[1];
  const int*   command= (const int*)  d_in[2];
  const float* sp_W1  = (const float*)d_in[3];  const float* sp_b1 = (const float*)d_in[4];
  const float* sp_W2  = (const float*)d_in[5];  const float* sp_b2 = (const float*)d_in[6];
  const float* sp_W3  = (const float*)d_in[7];  const float* sp_b3 = (const float*)d_in[8];
  const float* vp_W1  = (const float*)d_in[9];  const float* vp_b1 = (const float*)d_in[10];
  const float* vp_W2  = (const float*)d_in[11]; const float* vp_b2 = (const float*)d_in[12];
  const float* vp_W3  = (const float*)d_in[13]; const float* vp_b3 = (const float*)d_in[14];
  const float* join_W = (const float*)d_in[15]; const float* join_b= (const float*)d_in[16];
  const float* c_W1   = (const float*)d_in[17]; const float* c_b1  = (const float*)d_in[18];
  const float* c_W2   = (const float*)d_in[19]; const float* c_b2  = (const float*)d_in[20];
  const float* c_W3   = (const float*)d_in[21]; const float* c_b3  = (const float*)d_in[22];

  char* ws = (char*)d_ws;
  size_t off = 0;
  auto alloc = [&](size_t bytes) -> void* {
    void* p = ws + off;
    off = (off + bytes + 255) & ~(size_t)255;
    return p;
  };
  _Float16* p_i_h  = (_Float16*)alloc((size_t)B * 512 * 2);  // f16 copy of p_i
  _Float16* h128a  = (_Float16*)alloc((size_t)B * 128 * 2);  // ping
  _Float16* h128b  = (_Float16*)alloc((size_t)B * 128 * 2);  // pong
  _Float16* v_emb  = (_Float16*)alloc((size_t)B * 128 * 2);  // sp-MLP output
  _Float16* joined = (_Float16*)alloc((size_t)B * 512 * 2);
  _Float16* ch1    = (_Float16*)alloc((size_t)4 * B * 256 * 2);
  _Float16* ch2    = (_Float16*)alloc((size_t)4 * B * 256 * 2);
  _Float16* spW2t  = (_Float16*)alloc((size_t)128 * 128 * 2);
  _Float16* spW3t  = (_Float16*)alloc((size_t)128 * 128 * 2);
  _Float16* vpW1t  = (_Float16*)alloc((size_t)128 * 512 * 2);
  _Float16* vpW2t  = (_Float16*)alloc((size_t)128 * 128 * 2);
  _Float16* joinWt = (_Float16*)alloc((size_t)512 * 640 * 2);
  _Float16* cW1t   = (_Float16*)alloc((size_t)4 * 256 * 512 * 2);
  _Float16* cW2t   = (_Float16*)alloc((size_t)4 * 256 * 256 * 2);
  (void)ws_size; (void)in_sizes; (void)n_in; (void)out_size;

  float* out_vp     = (float*)d_out;       // [B,1]
  float* out_action = (float*)d_out + B;   // [B,3]

  // --- pack: p_i -> f16, weights -> f16 transposed [N,K] ---
  {
    int n = B * 512;
    k_cvt_f16<<<(n + 255) / 256, 256, 0, stream>>>(p_i, p_i_h, n);
  }
  auto tpose = [&](const float* s, _Float16* d, int K, int N, int E) {
    dim3 g((K * N + 255) / 256, E, 1);
    k_transpose_f16<<<g, 256, 0, stream>>>(s, d, K, N);
  };
  tpose(sp_W2, spW2t, 128, 128, 1);
  tpose(sp_W3, spW3t, 128, 128, 1);
  tpose(vp_W1, vpW1t, 512, 128, 1);
  tpose(vp_W2, vpW2t, 128, 128, 1);
  tpose(join_W, joinWt, 640, 512, 1);
  tpose(c_W1, cW1t, 512, 256, 4);
  tpose(c_W2, cW2t, 256, 256, 4);

  auto gemm = [&](const _Float16* A, size_t a_estride, int Ka,
                  const _Float16* A2, int Kb,
                  const _Float16* Bt, const float* bias, _Float16* Out,
                  int N, int relu, int E) {
    dim3 g(B / 256, N / 64, E);
    k_gemm_wmma<<<g, 256, 0, stream>>>(A, a_estride, Ka, A2, Kb, Bt, bias, Out, B, N, relu);
  };

  // --- measured-speed MLP: 1 -> 128 -> 128 -> 128 ---
  k_sp_layer1<<<(B * 128) / 256, 256, 0, stream>>>(speed, sp_W1, sp_b1, h128a);
  gemm(h128a, 0, 128, nullptr, 0, spW2t, sp_b2, h128b, 128, 1, 1);
  gemm(h128b, 0, 128, nullptr, 0, spW3t, sp_b3, v_emb, 128, 0, 1);

  // --- speed prediction: 512 -> 128 -> 128 -> 1 ---
  gemm(p_i_h, 0, 512, nullptr, 0, vpW1t, vp_b1, h128a, 128, 1, 1);
  gemm(h128a, 0, 128, nullptr, 0, vpW2t, vp_b2, h128b, 128, 1, 1);
  k_vp_head<<<B / 8, 256, 0, stream>>>(h128b, vp_W3, vp_b3, out_vp);

  // --- join: concat(p_i, v) [B,640] @ [640,512] (split-A, no concat materialized) ---
  gemm(p_i_h, 0, 512, v_emb, 128, joinWt, join_b, joined, 512, 0, 1);

  // --- control heads: 4 experts batched over grid.z ---
  gemm(joined, 0, 512, nullptr, 0, cW1t, c_b1, ch1, 256, 1, 4);
  gemm(ch1, (size_t)B * 256, 256, nullptr, 0, cW2t, c_b2, ch2, 256, 1, 4);

  // --- routed final projection (only selected expert) ---
  k_action_head<<<B / 8, 256, 0, stream>>>(ch2, command, c_W3, c_b3, out_action);
}
